// MambaBB_77541339562174
// MI455X (gfx1250) — compile-verified, matrix operand order checked by reading, not run on hardware
//
#include <hip/hip_runtime.h>
#include <hip/hip_bf16.h>
#include <math.h>

// ---------------------------------------------------------------------------
// MambaBB relational transformer for gfx1250 (MI455X).
// All GEMMs lowered to v_wmma_f32_16x16x32_f16 (f16 in, f32 accumulate).
// Key algebraic fold: rel_bias(adW[l]) is only consumed through biasW[l],
// so we precompute M_l = adW[l] @ biasW_l (128x8) on device and never
// materialize the (BT,N,N,512) bias tensors (43 GFLOP -> 0.13 GFLOP).
// GEMM tile count is a template parameter so the K-loop is branch-free
// (no conditional WMMA -> no accumulator shadow-copy v_mov storms).
// ---------------------------------------------------------------------------

typedef __attribute__((ext_vector_type(16))) _Float16 v16h;
typedef __attribute__((ext_vector_type(8)))  _Float16 v8h;
typedef __attribute__((ext_vector_type(8)))  float    v8f;

#define SHUF16(lo, hi) __builtin_shufflevector((lo), (hi), 0,1,2,3,4,5,6,7,8,9,10,11,12,13,14,15)

// Problem constants (from setup_inputs)
enum { Bb = 2, Tt = 128, Nn = 16, Din = 32, Dd = 512, Hh = 8, Ll = 4, Aa = 12 };
enum { BT = Bb * Tt };            // 256 frames
enum { ROWS = BT * Nn };          // 4096 token rows
enum { PAIRS = BT * Nn * Nn };    // 65536 pair rows
enum { HD = Dd / Hh };            // 64 head dim

// ---------------------------------------------------------------------------
// Generic WMMA GEMM:  C(MxN) = A(MxK,f16) * Bt(NxK,f16)^T + bias + res
// 128 threads = 4 waves; block computes 64x(NT*16) tile; each wave does
// 16 rows x NT col-tiles (A-fragment reused across the NT WMMAs).
// NT is compile-time: the K-loop is branch-free pure load+wmma.
// Epilogue: optional f32 store (with accumulate) and/or f16 store.
// ---------------------------------------------------------------------------
template <int NT>
__global__ void __launch_bounds__(128)
gemm_wmma_kernel(const _Float16* __restrict__ A, int lda,
                 const _Float16* __restrict__ Bt, int ldb,
                 const float* __restrict__ bias,
                 float* Cf32, int ldc, int accum,
                 _Float16* Cf16, int ldh,
                 const float* __restrict__ res, int ldr,
                 int M, int N, int K)
{
    const int lane = threadIdx.x & 31;
    const int wave = threadIdx.x >> 5;
    const int lr   = lane & 15;       // row/col within fragment
    const int hs   = lane >> 4;       // half-select (K offset group)
    const int m0   = blockIdx.x * 64 + wave * 16;
    const int n0   = blockIdx.y * (NT * 16);
    if (m0 >= M) return;

    v8f acc[NT] = {};

    const _Float16* arow  = A  + (size_t)(m0 + lr) * lda + hs * 8;
    const _Float16* brow0 = Bt + (size_t)(n0 + lr) * ldb + hs * 8;

    for (int k = 0; k < K; k += 32) {
        v8h alo = *(const v8h*)(arow + k);
        v8h ahi = *(const v8h*)(arow + k + 16);
        v16h a  = SHUF16(alo, ahi);
        if (k + 128 < K) __builtin_prefetch(arow + k + 128, 0, 0); // global_prefetch_b8
#pragma unroll
        for (int t = 0; t < NT; ++t) {
            const _Float16* br = brow0 + (size_t)(t * 16) * ldb + k;
            v8h blo = *(const v8h*)(br);
            v8h bhi = *(const v8h*)(br + 16);
            v16h b  = SHUF16(blo, bhi);
            acc[t] = __builtin_amdgcn_wmma_f32_16x16x32_f16(
                false, a, false, b, (short)0, acc[t], false, false);
        }
    }

#pragma unroll
    for (int t = 0; t < NT; ++t) {
        const int nc = n0 + t * 16 + lr;
#pragma unroll
        for (int r = 0; r < 8; ++r) {
            if (nc < N) {
                const int rg = m0 + hs * 8 + r;   // C layout: row = vgpr + 8*hs
                float v = acc[t][r];
                if (bias) v += bias[nc];
                if (res)  v += res[(size_t)rg * ldr + nc];
                if (Cf32) {
                    size_t ix = (size_t)rg * ldc + nc;
                    Cf32[ix] = accum ? (Cf32[ix] + v) : v;
                }
                if (Cf16) Cf16[(size_t)rg * ldh + nc] = (_Float16)v;
            }
        }
    }
}

// ---------------------------------------------------------------------------
// Per-(frame,head) attention: q,k,v are 16x64 slices of the f16 qkv buffer.
// scores = q k^T * 0.125 + b ; softmax over lanes ; out = P v via WMMA with
// P re-striped C-layout -> A-layout through LDS (zero-padded K=16 -> 32).
// One wave per (frame, head); 4 waves per block; grid = BT*H/4 exactly.
// ---------------------------------------------------------------------------
__global__ void __launch_bounds__(128)
attn_wmma_kernel(const _Float16* __restrict__ qkv,   // (ROWS, 1536) f16
                 const float* __restrict__ bvec,     // (PAIRS, 8) f32
                 _Float16* __restrict__ out)         // (ROWS, 512) f16
{
    __shared__ __align__(16) _Float16 plds[4][16 * 16];
    const int wave = threadIdx.x >> 5;
    const int lane = threadIdx.x & 31;
    const int lr   = lane & 15;
    const int hs   = lane >> 4;
    const int gw   = blockIdx.x * 4 + wave;
    const int f    = gw >> 3;          // frame
    const int h    = gw & 7;           // head

    const _Float16* qrow = qkv + (size_t)(f * 16 + lr) * (3 * Dd) + h * HD + hs * 8;
    const _Float16* krow = qrow + Dd;

    v8f s = {};
#pragma unroll
    for (int k = 0; k < HD; k += 32) {
        v8h alo = *(const v8h*)(qrow + k);
        v8h ahi = *(const v8h*)(qrow + k + 16);
        v8h blo = *(const v8h*)(krow + k);
        v8h bhi = *(const v8h*)(krow + k + 16);
        v16h a = SHUF16(alo, ahi);
        v16h b = SHUF16(blo, bhi);
        s = __builtin_amdgcn_wmma_f32_16x16x32_f16(false, a, false, b, (short)0, s, false, false);
    }

    // scale + relational bias, then row softmax (row m lives across 16 lanes
    // of one half-wave; xor masks 1..8 stay within the half).
    float p[8];
#pragma unroll
    for (int r = 0; r < 8; ++r) {
        const int m = hs * 8 + r;
        p[r] = s[r] * 0.125f + bvec[((size_t)(f * 16 + m) * 16 + lr) * 8 + h];
    }
#pragma unroll
    for (int r = 0; r < 8; ++r) {
        float mx = p[r];
        for (int d = 1; d < 16; d <<= 1) mx = fmaxf(mx, __shfl_xor(mx, d, 32));
        float e = __expf(p[r] - mx);
        float sm = e;
        for (int d = 1; d < 16; d <<= 1) sm += __shfl_xor(sm, d, 32);
        p[r] = e / sm;
    }

    // C-layout -> LDS (row-major 16x16) -> A-layout fragment
#pragma unroll
    for (int r = 0; r < 8; ++r) {
        const int m = hs * 8 + r;
        plds[wave][m * 16 + lr] = (_Float16)p[r];
    }
    __syncthreads();
    v8h plo = *(const v8h*)&plds[wave][lr * 16 + hs * 8];
    v8h zz  = {};
    v16h pa = SHUF16(plo, zz);   // K=16..31 zero-padded

    // out = P (16x16, padded to 16x32) @ v (16x64) in 4 column chunks
    const _Float16* vbase = qkv + 2 * Dd + h * HD;
#pragma unroll
    for (int c = 0; c < 4; ++c) {
        v16h bv = {};
#pragma unroll
        for (int e = 0; e < 8; ++e) {
            const int kk = hs * 8 + e;   // agent index (Bt row = v^T column)
            bv[e] = vbase[(size_t)(f * 16 + kk) * (3 * Dd) + c * 16 + lr];
        }
        v8f o = {};
        o = __builtin_amdgcn_wmma_f32_16x16x32_f16(false, pa, false, bv, (short)0, o, false, false);
#pragma unroll
        for (int r = 0; r < 8; ++r) {
            const int m = hs * 8 + r;
            out[(size_t)(f * 16 + m) * Dd + h * HD + c * 16 + lr] = (_Float16)o[r];
        }
    }
}

// ---------------------------------------------------------------------------
// Row ops: one wave per row of width D.
// mode 0: out = rmsnorm(x,w) ; mode 1: out = silu(rmsnorm(x,w)) ;
// mode 2: out = x + rmsnorm(x,w) (in-place capable)
// ---------------------------------------------------------------------------
__global__ void __launch_bounds__(128)
rowops_kernel(const float* __restrict__ x, const float* __restrict__ w,
              float* of32, _Float16* of16, int ldo16, int M, int D, int mode)
{
    const int row  = blockIdx.x * 4 + (threadIdx.x >> 5);
    const int lane = threadIdx.x & 31;
    if (row >= M) return;
    const float* xr = x + (size_t)row * D;
    float ss = 0.f;
    for (int j = lane; j < D; j += 32) { float v = xr[j]; ss += v * v; }
    for (int d = 16; d; d >>= 1) ss += __shfl_xor(ss, d, 32);
    const float sc = rsqrtf(ss / (float)D + 1e-5f);
    for (int j = lane; j < D; j += 32) {
        float v = xr[j];
        float y = v * sc * w[j];
        if (mode == 1) y = y / (1.f + __expf(-y));   // silu
        if (mode == 2) y = v + y;                    // residual add-norm
        if (of32) of32[(size_t)row * D + j] = y;
        if (of16) of16[(size_t)row * ldo16 + j] = (_Float16)y;
    }
}

// Pairwise geometric features (11 real + pad to 32 for WMMA K-step)
__global__ void feat_kernel(const float* __restrict__ pos, const float* __restrict__ vel,
                            _Float16* __restrict__ feat)
{
    const int pid = blockIdx.x * blockDim.x + threadIdx.x;
    if (pid >= PAIRS) return;
    const int n  = pid & 15;
    const int m  = (pid >> 4) & 15;
    const int fr = pid >> 8;
    const float* pm = pos + (size_t)(fr * 16 + m) * 2;
    const float* pn = pos + (size_t)(fr * 16 + n) * 2;
    const float* vm = vel + (size_t)(fr * 16 + m) * 2;
    const float* vn = vel + (size_t)(fr * 16 + n) * 2;
    float dx = pm[0] - pn[0]; dx -= rintf(dx * (1.f / 1024.f)) * 1024.f;
    float dy = pm[1] - pn[1]; dy -= rintf(dy * (1.f / 1024.f)) * 1024.f;
    float dvx = vm[0] - vn[0], dvy = vm[1] - vn[1];
    float dist = sqrtf(dx * dx + dy * dy) + 1e-6f;
    float invd = 1.f / (dist + 0.1f);
    float rs   = sqrtf(dvx * dvx + dvy * dvy);
    float clo  = (dvx * dx + dvy * dy) / dist;
    float f[11] = { dx, dy, dvx, dvy, dist, invd, rs, clo, dx / dist, dy / dist, logf(dist + 1.f) };
    _Float16* o = feat + (size_t)pid * 32;
#pragma unroll
    for (int j = 0; j < 32; ++j) o[j] = (j < 11) ? (_Float16)f[j] : (_Float16)0.f;
}

// Fold adW[l] @ biasW_l -> Mt (5 x 16 x 128, f16, padded heads) and
// adB[l] @ biasW_l -> ob (5 x 16, f32)
__global__ void mkbias_kernel(const float* __restrict__ adW, const float* __restrict__ adB,
                              const float* __restrict__ abiasW, const float* __restrict__ biasW,
                              _Float16* __restrict__ Mt, float* __restrict__ ob)
{
    const int tid = blockIdx.x * blockDim.x + threadIdx.x;
    if (tid >= 5 * 16 * 128) return;
    const int c = tid & 127;
    const int h = (tid >> 7) & 15;
    const int l = tid >> 11;
    const float* bw = (l == 0) ? abiasW : (biasW + (size_t)(l - 1) * Dd * Hh);
    float acc = 0.f;
    if (h < Hh) {
        const float* wl = adW + (size_t)l * 128 * Dd + (size_t)c * Dd;
        for (int j = 0; j < Dd; ++j) acc += wl[j] * bw[j * Hh + h];
    }
    Mt[((size_t)l * 16 + h) * 128 + c] = (_Float16)acc;
    if (c == 0) {
        float o = 0.f;
        if (h < Hh) for (int j = 0; j < Dd; ++j) o += adB[(size_t)l * Dd + j] * bw[j * Hh + h];
        ob[l * 16 + h] = o;
    }
}

// f32 weight (K x N) -> f16 transposed (Npad x Kpad), zero padded
__global__ void convT_kernel(const float* __restrict__ W, _Float16* __restrict__ out,
                             int K, int N, int Kpad, int Npad)
{
    const int tid = blockIdx.x * blockDim.x + threadIdx.x;
    if (tid >= Npad * Kpad) return;
    const int k = tid % Kpad;
    const int n = tid / Kpad;
    out[tid] = (k < K && n < N) ? (_Float16)W[(size_t)k * N + n] : (_Float16)0.f;
}

__global__ void cvt16_kernel(const float* __restrict__ in, _Float16* __restrict__ out, int n)
{
    const int tid = blockIdx.x * blockDim.x + threadIdx.x;
    if (tid < n) out[tid] = (_Float16)in[tid];
}

// action embeddings -> columns 512..639 of the concat buffer
__global__ void aemb_kernel(const int* __restrict__ pact,
                            const float* __restrict__ embP, const float* __restrict__ embT,
                            const float* __restrict__ embS, _Float16* __restrict__ xcat)
{
    const int tid = blockIdx.x * blockDim.x + threadIdx.x;
    if (tid >= ROWS * 128) return;
    const int row = tid >> 7;
    const int j   = tid & 127;
    const int a0 = pact[row * 3 + 0], a1 = pact[row * 3 + 1], a2 = pact[row * 3 + 2];
    float v = (j < 32) ? embP[a0 * 32 + j]
            : (j < 96) ? embT[a1 * 64 + (j - 32)]
                       : embS[a2 * 32 + (j - 96)];
    xcat[(size_t)row * 640 + 512 + j] = (_Float16)v;
}

// ---------------------------------------------------------------------------
extern "C" void kernel_launch(void* const* d_in, const int* in_sizes, int n_in,
                              void* d_out, int out_size, void* d_ws, size_t ws_size,
                              hipStream_t stream)
{
    const float* state  = (const float*)d_in[0];
    const int*   pact   = (const int*)  d_in[1];
    const float* pos    = (const float*)d_in[2];
    const float* vel    = (const float*)d_in[3];
    const float* seW1   = (const float*)d_in[4];
    const float* seB1   = (const float*)d_in[5];
    const float* seN    = (const float*)d_in[6];
    const float* seW2   = (const float*)d_in[7];
    const float* seB2   = (const float*)d_in[8];
    const float* tW1    = (const float*)d_in[9];
    const float* tB1    = (const float*)d_in[10];
    const float* tN     = (const float*)d_in[11];
    const float* tW2    = (const float*)d_in[12];
    const float* tB2    = (const float*)d_in[13];
    const float* adW    = (const float*)d_in[14];
    const float* adB    = (const float*)d_in[15];
    const float* aqkvW  = (const float*)d_in[16];
    const float* aqkvB  = (const float*)d_in[17];
    const float* aprojW = (const float*)d_in[18];
    const float* aprojB = (const float*)d_in[19];
    const float* abiasW = (const float*)d_in[20];
    const float* ahW    = (const float*)d_in[21];
    const float* ahB    = (const float*)d_in[22];
    const float* embP   = (const float*)d_in[23];
    const float* embT   = (const float*)d_in[24];
    const float* embS   = (const float*)d_in[25];
    const float* fW     = (const float*)d_in[26];
    const float* fB     = (const float*)d_in[27];
    const float* n1     = (const float*)d_in[28];
    const float* n2     = (const float*)d_in[29];
    const float* qkvW   = (const float*)d_in[30];
    const float* qkvB   = (const float*)d_in[31];
    const float* projW  = (const float*)d_in[32];
    const float* projB  = (const float*)d_in[33];
    const float* biasW  = (const float*)d_in[34];
    const float* whW    = (const float*)d_in[35];
    const float* whB    = (const float*)d_in[36];
    float* outF = (float*)d_out;   // [state_pred (4096*32) | action_logits (4096*12)]

    // ---- workspace layout ----
    char* wsb = (char*)d_ws;
    size_t off = 0;
    auto alloc = [&](size_t bytes) -> char* {
        char* p = wsb + off;
        off = (off + bytes + 255) & ~(size_t)255;
        return p;
    };
    _Float16* seW1t   = (_Float16*)alloc((size_t)512 * 32 * 2);
    _Float16* seW2t   = (_Float16*)alloc((size_t)512 * 512 * 2);
    _Float16* tW1t    = (_Float16*)alloc((size_t)128 * 32 * 2);
    _Float16* tW2t    = (_Float16*)alloc((size_t)128 * 128 * 2);
    _Float16* aqkvWt  = (_Float16*)alloc((size_t)1536 * 512 * 2);
    _Float16* aprojWt = (_Float16*)alloc((size_t)512 * 512 * 2);
    _Float16* fWt     = (_Float16*)alloc((size_t)512 * 640 * 2);
    _Float16* qkvWt   = (_Float16*)alloc((size_t)4 * 1536 * 512 * 2);
    _Float16* projWt  = (_Float16*)alloc((size_t)4 * 512 * 512 * 2);
    _Float16* whWt    = (_Float16*)alloc((size_t)32 * 512 * 2);
    _Float16* ahWt    = (_Float16*)alloc((size_t)16 * 512 * 2);
    _Float16* Mt      = (_Float16*)alloc((size_t)5 * 16 * 128 * 2);
    float*    ob      = (float*)   alloc((size_t)5 * 16 * 4);
    _Float16* feat    = (_Float16*)alloc((size_t)PAIRS * 32 * 2);
    float*    trkpre  = (float*)   alloc((size_t)PAIRS * 128 * 4);
    _Float16* trkact  = (_Float16*)alloc((size_t)PAIRS * 128 * 2);
    _Float16* trkout  = (_Float16*)alloc((size_t)PAIRS * 128 * 2);
    float*    bb      = (float*)   alloc((size_t)5 * PAIRS * 8 * 4);
    _Float16* st16    = (_Float16*)alloc((size_t)ROWS * 32 * 2);
    float*    pre     = (float*)   alloc((size_t)ROWS * 512 * 4);
    _Float16* xcat    = (_Float16*)alloc((size_t)ROWS * 640 * 2);
    _Float16* act2    = (_Float16*)alloc((size_t)ROWS * 512 * 2);
    _Float16* qkv16   = (_Float16*)alloc((size_t)ROWS * 1536 * 2);
    _Float16* attn16  = (_Float16*)alloc((size_t)ROWS * 512 * 2);
    _Float16* xact16  = (_Float16*)alloc((size_t)ROWS * 512 * 2);
    float*    xbuf    = (float*)   alloc((size_t)ROWS * 512 * 4);

    // Dispatch with compile-time tile count: branch-free WMMA K-loop.
    auto gemm = [&](const _Float16* A, int lda, const _Float16* Bt, int ldb,
                    const float* bias, float* C32, int ldc, int accum,
                    _Float16* C16, int ldh, const float* res, int ldr,
                    int M, int N, int K) {
        const int nt = (N >= 64) ? 4 : ((N + 15) / 16);   // uniform per launch
        dim3 g((M + 63) / 64, (N + nt * 16 - 1) / (nt * 16));
        switch (nt) {
        case 1: gemm_wmma_kernel<1><<<g, 128, 0, stream>>>(A, lda, Bt, ldb, bias, C32, ldc, accum, C16, ldh, res, ldr, M, N, K); break;
        case 2: gemm_wmma_kernel<2><<<g, 128, 0, stream>>>(A, lda, Bt, ldb, bias, C32, ldc, accum, C16, ldh, res, ldr, M, N, K); break;
        case 3: gemm_wmma_kernel<3><<<g, 128, 0, stream>>>(A, lda, Bt, ldb, bias, C32, ldc, accum, C16, ldh, res, ldr, M, N, K); break;
        default: gemm_wmma_kernel<4><<<g, 128, 0, stream>>>(A, lda, Bt, ldb, bias, C32, ldc, accum, C16, ldh, res, ldr, M, N, K); break;
        }
    };
    auto convT = [&](const float* W, _Float16* o, int K, int N, int Kp, int Np) {
        convT_kernel<<<(Np * Kp + 255) / 256, 256, 0, stream>>>(W, o, K, N, Kp, Np);
    };

    // ---- weight prep (f16, transposed, padded) ----
    convT(seW1, seW1t, 32, 512, 32, 512);
    convT(seW2, seW2t, 512, 512, 512, 512);
    convT(tW1,  tW1t,  12, 128, 32, 128);
    convT(tW2,  tW2t,  128, 128, 128, 128);
    convT(aqkvW, aqkvWt, 512, 1536, 512, 1536);
    convT(aprojW, aprojWt, 512, 512, 512, 512);
    convT(fW, fWt, 640, 512, 640, 512);
    for (int i = 0; i < Ll; ++i) {
        convT(qkvW + (size_t)i * 512 * 1536, qkvWt + (size_t)i * 1536 * 512, 512, 1536, 512, 1536);
        convT(projW + (size_t)i * 512 * 512, projWt + (size_t)i * 512 * 512, 512, 512, 512, 512);
    }
    convT(whW, whWt, 512, 32, 512, 32);
    convT(ahW, ahWt, 512, 12, 512, 16);
    mkbias_kernel<<<(5 * 16 * 128 + 255) / 256, 256, 0, stream>>>(adW, adB, abiasW, biasW, Mt, ob);

    // ---- relational trunk (pair features) ----
    feat_kernel<<<PAIRS / 256, 256, 0, stream>>>(pos, vel, feat);
    gemm(feat, 32, tW1t, 32, tB1, trkpre, 128, 0, nullptr, 0, nullptr, 0, PAIRS, 128, 32);
    rowops_kernel<<<(PAIRS + 3) / 4, 128, 0, stream>>>(trkpre, tN, nullptr, trkact, 128, PAIRS, 128, 1);
    gemm(trkact, 128, tW2t, 128, tB2, nullptr, 0, 0, trkout, 128, nullptr, 0, PAIRS, 128, 128);
    for (int l = 0; l < 5; ++l)   // folded relational bias per attention use
        gemm(trkout, 128, Mt + (size_t)l * 16 * 128, 128, ob + l * 16,
             bb + (size_t)l * PAIRS * 8, 8, 0, nullptr, 0, nullptr, 0, PAIRS, 8, 128);

    // ---- state encoder -> s_emb (f16, written into concat cols 0..511) ----
    cvt16_kernel<<<(ROWS * 32 + 255) / 256, 256, 0, stream>>>(state, st16, ROWS * 32);
    gemm(st16, 32, seW1t, 32, seB1, pre, 512, 0, nullptr, 0, nullptr, 0, ROWS, 512, 32);
    rowops_kernel<<<ROWS / 4, 128, 0, stream>>>(pre, seN, nullptr, act2, 512, ROWS, 512, 1);
    gemm(act2, 512, seW2t, 512, seB2, nullptr, 0, 0, xcat, 640, nullptr, 0, ROWS, 512, 512);
    aemb_kernel<<<(ROWS * 128 + 255) / 256, 256, 0, stream>>>(pact, embP, embT, embS, xcat);

    // ---- actor branch ----
    gemm(xcat, 640, aqkvWt, 512, aqkvB, nullptr, 0, 0, qkv16, 1536, nullptr, 0, ROWS, 1536, 512);
    attn_wmma_kernel<<<BT * Hh / 4, 128, 0, stream>>>(qkv16, bb, attn16);
    gemm(attn16, 512, aprojWt, 512, aprojB, nullptr, 0, 0, xact16, 512, nullptr, 0, ROWS, 512, 512);
    gemm(xact16, 512, ahWt, 512, ahB, outF + ROWS * 32, 12, 0, nullptr, 0, nullptr, 0, ROWS, 12, 512);

    // ---- fusion + L spatial-attention layers (mamba = identity + skip) ----
    gemm(xcat, 640, fWt, 640, fB, xbuf, 512, 0, nullptr, 0, nullptr, 0, ROWS, 512, 640);
    for (int i = 0; i < Ll; ++i) {
        rowops_kernel<<<ROWS / 4, 128, 0, stream>>>(xbuf, n1 + (size_t)i * 512, xbuf, nullptr, 0, ROWS, 512, 2);
        rowops_kernel<<<ROWS / 4, 128, 0, stream>>>(xbuf, n2 + (size_t)i * 512, nullptr, act2, 512, ROWS, 512, 0);
        gemm(act2, 512, qkvWt + (size_t)i * 1536 * 512, 512, qkvB + (size_t)i * 1536,
             nullptr, 0, 0, qkv16, 1536, nullptr, 0, ROWS, 1536, 512);
        attn_wmma_kernel<<<BT * Hh / 4, 128, 0, stream>>>(qkv16, bb + (size_t)(i + 1) * PAIRS * 8, attn16);
        gemm(attn16, 512, projWt + (size_t)i * 512 * 512, 512, projB + (size_t)i * 512,
             xbuf, 512, /*accum=*/1, nullptr, 0, nullptr, 0, ROWS, 512, 512);
    }

    // ---- world-model head: state_pred = state + x @ whW + whB ----
    cvt16_kernel<<<(ROWS * 512 + 255) / 256, 256, 0, stream>>>(xbuf, act2, ROWS * 512);
    gemm(act2, 512, whWt, 512, whB, outF, 32, 0, nullptr, 0, state, 32, ROWS, 32, 512);
}
